// MultiHeadAttention_18622978195667
// MI455X (gfx1250) — matrix operand order repaired and checked by reading, not compiled, not run on hardware
//
#include <hip/hip_runtime.h>
#include <hip/hip_bf16.h>

// ---------------- problem constants ----------------
#define DM   1024          // d_model
#define NH   16            // heads
#define DK   64            // head dim
#define BB   2             // batch
#define SS   2048          // seq len
#define NROW (BB*SS)       // 4096 flattened rows

typedef __attribute__((ext_vector_type(16))) __bf16 v16bf;
typedef __attribute__((ext_vector_type(8)))  __bf16 v8bf;
typedef __attribute__((ext_vector_type(8)))  float  v8f;
typedef __attribute__((ext_vector_type(4)))  unsigned int v4u;
typedef __attribute__((ext_vector_type(8)))  int v8i_t;
typedef __attribute__((ext_vector_type(4)))  int v4i_t;

static __device__ __forceinline__ v8f wmma_bf16(v16bf a, v16bf b, v8f c) {
  // 8 args: (neg_a, A, neg_b, B, c_mod, C, reuse_a, reuse_b)
  return __builtin_amdgcn_wmma_f32_16x16x32_bf16(
      false, a, false, b, (short)0, c, false, false);
}

static __device__ __forceinline__ v8f vzero8() {
  v8f z = {0.f,0.f,0.f,0.f,0.f,0.f,0.f,0.f};
  return z;
}

// Build A fragment (16x32 bf16) from a row-major bf16 row pointer.
// lane<16: elems 0..15 = K 0..15 (row m = lane)
// lane>=16: elems 0..15 = K 16..31 (row m = lane-16)
static __device__ __forceinline__ v16bf load_a_frag(const __bf16* row, int lane) {
  const int klo = (lane < 16) ? 0 : 16;
  v8bf lo = *(const v8bf*)(row + klo);
  v8bf hi = *(const v8bf*)(row + 8 + klo);
  v16bf a;
#pragma unroll
  for (int i = 0; i < 8; ++i) { a[i] = lo[i]; a[i+8] = hi[i]; }
  return a;
}

// Build B fragment (32x16 bf16) where B[k][n] is contiguous in k for fixed n
// (i.e. source is row-major over k for "row n"). lane: n = lane%16,
// elems i = k = 16*(lane/16) + i.
static __device__ __forceinline__ v16bf load_b_frag_rowk(const __bf16* nrow, int lane) {
  const int kb = (lane >> 4) * 16;
  v8bf lo = *(const v8bf*)(nrow + kb);
  v8bf hi = *(const v8bf*)(nrow + kb + 8);
  v16bf b;
#pragma unroll
  for (int i = 0; i < 8; ++i) { b[i] = lo[i]; b[i+8] = hi[i]; }
  return b;
}

// =====================================================================
// Projection GEMM: Y = X @ W^T + bias
//   X : (NROW, DM) fp32, W : (DM, DM) fp32 row-major, bias : (DM)
//   HEAD_SPLIT=true : write bf16 head-split (B,H,S,DK)
//   HEAD_SPLIT=false: write fp32 flat (B,S,D)
// Block = 128 threads (4 waves); 64x64 output tile; wave w owns rows 16w..16w+15.
// =====================================================================
template <bool HEAD_SPLIT>
__global__ __launch_bounds__(128)
void proj_kernel(const float* __restrict__ X,
                 const float* __restrict__ W,
                 const float* __restrict__ bias,
                 __bf16* __restrict__ out_bf,
                 float*  __restrict__ out_f) {
  __shared__ __align__(16) __bf16 Xs[64 * 32];
  __shared__ __align__(16) __bf16 Ws[64 * 32];

  const int tid  = threadIdx.x;
  const int lane = tid & 31;
  const int wave = tid >> 5;
  const int row0 = blockIdx.y * 64;
  const int col0 = blockIdx.x * 64;

  v8f acc[4];
#pragma unroll
  for (int i = 0; i < 4; ++i) acc[i] = vzero8();

  for (int ks = 0; ks < DM; ks += 32) {
    // Stage X tile (64x32) and W tile (64x32) as bf16 into LDS.
#pragma unroll
    for (int i = 0; i < 16; ++i) {
      const int idx = tid * 16 + i;           // 0..2047
      const int r = idx >> 5, c = idx & 31;
      Xs[idx] = (__bf16)X[(size_t)(row0 + r) * DM + ks + c];
      Ws[idx] = (__bf16)W[(size_t)(col0 + r) * DM + ks + c];
    }
    if (ks + 32 < DM) {  // prefetch next slab -> global_prefetch_b8
      __builtin_prefetch(&X[(size_t)(row0 + (tid >> 1)) * DM + ks + 32], 0, 3);
      __builtin_prefetch(&W[(size_t)(col0 + (tid >> 1)) * DM + ks + 32], 0, 3);
    }
    __syncthreads();

    const v16bf a = load_a_frag(&Xs[(16 * wave + (lane & 15)) * 32], lane);
#pragma unroll
    for (int cb = 0; cb < 4; ++cb) {
      const int nloc = cb * 16 + (lane & 15);
      const v16bf b = load_b_frag_rowk(&Ws[nloc * 32], lane);
      acc[cb] = wmma_bf16(a, b, acc[cb]);
    }
    __syncthreads();
  }

  // Epilogue: bias add + store. D layout: lane L, elem v -> M = v + 8*(L/16), N = L%16.
#pragma unroll
  for (int cb = 0; cb < 4; ++cb) {
    const int n  = col0 + cb * 16 + (lane & 15);
    const float bv = bias[n];
#pragma unroll
    for (int v = 0; v < 8; ++v) {
      const int m = row0 + wave * 16 + v + ((lane >> 4) << 3);
      const float val = acc[cb][v] + bv;
      if constexpr (HEAD_SPLIT) {
        const int b = m / SS, s = m % SS;
        const int h = n / DK, d = n % DK;
        out_bf[((((size_t)b * NH + h) * SS) + s) * DK + d] = (__bf16)val;
      } else {
        out_f[(size_t)m * DM + n] = val;
      }
    }
  }
}

// =====================================================================
// Attention scores + softmax -> attn_weights (fp32, (B,H,S,S)).
// One wave per (bh, 16-query block). Two passes over keys:
//   pass 1: online per-lane (max, sumexp), then xor-reduce across 16 lanes
//   pass 2: recompute QK^T, write normalized probabilities.
// Recomputation trades ~17 GF (free) for not spilling 537 MB of raw scores.
// =====================================================================
__global__ __launch_bounds__(128)
void attn_scores_kernel(const __bf16* __restrict__ q_ws,
                        const __bf16* __restrict__ k_ws,
                        float* __restrict__ attn_w) {
  const int lane = threadIdx.x & 31;
  const int wave = threadIdx.x >> 5;
  const int task = blockIdx.x * 4 + wave;     // 0 .. B*H*(S/16)-1
  const int bh   = task >> 7;                 // / (S/16)
  const int qblk = task & 127;

  const __bf16* qbase = q_ws + (size_t)bh * SS * DK;
  const __bf16* kbase = k_ws + (size_t)bh * SS * DK;
  float* pout = attn_w + (size_t)bh * SS * SS;

  // Q fragments for the two K=32 steps of dk=64.
  v16bf aq[2];
  {
    const int m = qblk * 16 + (lane & 15);
#pragma unroll
    for (int s = 0; s < 2; ++s)
      aq[s] = load_a_frag(&qbase[(size_t)m * DK + 32 * s], lane);
  }

  const float scale = 0.125f;  // 1/sqrt(64)
  float mrun[8], srun[8];
#pragma unroll
  for (int v = 0; v < 8; ++v) { mrun[v] = -3.0e38f; srun[v] = 0.f; }

  // ---- pass 1: online max / sum-exp ----
  for (int nb = 0; nb < SS / 16; ++nb) {
    v8f acc = vzero8();
    const int n = nb * 16 + (lane & 15);
#pragma unroll
    for (int s = 0; s < 2; ++s) {
      const v16bf b = load_b_frag_rowk(&kbase[(size_t)n * DK + 32 * s], lane);
      acc = wmma_bf16(aq[s], b, acc);
    }
#pragma unroll
    for (int v = 0; v < 8; ++v) {
      const float x  = acc[v] * scale;
      const float nm = fmaxf(mrun[v], x);
      srun[v] = srun[v] * __expf(mrun[v] - nm) + __expf(x - nm);
      mrun[v] = nm;
    }
  }

  // ---- reduce (m,s) across the 16 lanes sharing each row ----
#pragma unroll
  for (int v = 0; v < 8; ++v) {
    float m = mrun[v], s = srun[v];
#pragma unroll
    for (int off = 1; off < 16; off <<= 1) {
      const float mo = __shfl_xor(m, off, 32);
      const float so = __shfl_xor(s, off, 32);
      const float nm = fmaxf(m, mo);
      s = s * __expf(m - nm) + so * __expf(mo - nm);
      m = nm;
    }
    mrun[v] = m;
    srun[v] = 1.0f / s;
  }

  // ---- pass 2: recompute and write normalized probabilities ----
  for (int nb = 0; nb < SS / 16; ++nb) {
    v8f acc = vzero8();
    const int n = nb * 16 + (lane & 15);
#pragma unroll
    for (int s = 0; s < 2; ++s) {
      const v16bf b = load_b_frag_rowk(&kbase[(size_t)n * DK + 32 * s], lane);
      acc = wmma_bf16(aq[s], b, acc);
    }
#pragma unroll
    for (int v = 0; v < 8; ++v) {
      const int row = qblk * 16 + v + ((lane >> 4) << 3);
      const float p = __expf(acc[v] * scale - mrun[v]) * srun[v];
      pout[(size_t)row * SS + n] = p;
    }
  }
}

// =====================================================================
// attn_out = P @ V : per (bh, 64-row group). P fp32 (B,H,S,S), V bf16
// (B,H,S,DK). The V slab (32k x 64n, 4 KB) is fetched by the Tensor Data
// Mover into LDS (wave 0 issues a 2-D D#; tile = 64 x 32 rows of bf16),
// fenced with s_wait_tensorcnt + workgroup barrier, shared by all 4 waves.
// Output fp32 merged-head layout (B,S,D).
// =====================================================================
__global__ __launch_bounds__(128)
void attn_pv_kernel(const float* __restrict__ P,
                    const __bf16* __restrict__ v_ws,
                    float* __restrict__ attn_out) {
  __shared__ __align__(128) __bf16 Vs[32 * DK];  // 4 KB

  const int tid  = threadIdx.x;
  const int lane = tid & 31;
  const int wave = tid >> 5;
  const int bh   = blockIdx.x >> 5;             // 32 row-groups per bh
  const int rg   = blockIdx.x & 31;
  const int b    = bh >> 4, h = bh & 15;

  const float* pbase  = P    + (size_t)bh * SS * SS;
  const __bf16* vbase = v_ws + (size_t)bh * SS * DK;

  // LDS byte offset of Vs (low 32 bits of the flat shared address).
  const unsigned lds_base = (unsigned)(size_t)&Vs[0];

  v8f acc[4];
#pragma unroll
  for (int i = 0; i < 4; ++i) acc[i] = vzero8();

  const int mrow = rg * 64 + wave * 16 + (lane & 15);

  for (int ks = 0; ks < SS; ks += 32) {
    // ---- TDM: DMA V[ks..ks+32)[0..64) bf16 into LDS (wave 0 only) ----
    if (wave == 0) {
      const unsigned long long ga =
          (unsigned long long)(size_t)(vbase + (size_t)ks * DK);
      v4u g0;
      g0[0] = 1u;                                       // count=1 (valid user D#)
      g0[1] = lds_base;                                 // lds_addr [63:32]
      g0[2] = (unsigned)(ga & 0xffffffffu);             // global_addr[31:0]
      g0[3] = (unsigned)((ga >> 32) & 0x01ffffffu)      // global_addr[56:32]
              | (2u << 30);                             // type=2 (image)
      v8i_t g1;
      g1[0] = (1 << 16);            // workgroup_mask=0, data_size=1 (2 bytes)
      g1[1] = (DK << 16);           // tensor_dim0[15:0]=64 in bits [63:48]
      g1[2] = (SS << 16);           // tensor_dim0 hi=0 | tensor_dim1[15:0]=2048
      g1[3] = (DK << 16);           // tensor_dim1 hi=0 | tile_dim0=64
      g1[4] = 32;                   // tile_dim1=32, tile_dim2=0
      g1[5] = DK;                   // tensor_dim0_stride=64 (low 32)
      g1[6] = 0;                    // stride0 hi | stride1 lo
      g1[7] = 0;                    // stride1 hi
      v4i_t zz = {0, 0, 0, 0};      // 2-D tensor: groups 2/3 unused
      v8i_t z8 = {0, 0, 0, 0, 0, 0, 0, 0};  // extra group (6-arg builtin form)
      __builtin_amdgcn_tensor_load_to_lds(g0, g1, zz, zz, z8, 0);
      __builtin_amdgcn_s_wait_tensorcnt(0);
    }
    __syncthreads();

    // A fragment from P, converted fp32 -> bf16 on load.
    v16bf a;
    {
      const int klo = (lane < 16) ? 0 : 16;
      const float* prow = &pbase[(size_t)mrow * SS + ks];
#pragma unroll
      for (int i = 0; i < 8; ++i) {
        a[i]     = (__bf16)prow[klo + i];
        a[i + 8] = (__bf16)prow[8 + klo + i];
      }
    }
#pragma unroll
    for (int cb = 0; cb < 4; ++cb) {
      const int n  = cb * 16 + (lane & 15);
      const int kb = (lane >> 4) * 16;
      v16bf bf;
#pragma unroll
      for (int i = 0; i < 16; ++i) bf[i] = Vs[(kb + i) * DK + n];
      acc[cb] = wmma_bf16(a, bf, acc[cb]);
    }
    __syncthreads();
  }

#pragma unroll
  for (int cb = 0; cb < 4; ++cb) {
    const int d = h * DK + cb * 16 + (lane & 15);
#pragma unroll
    for (int v = 0; v < 8; ++v) {
      const int srow = rg * 64 + wave * 16 + v + ((lane >> 4) << 3);
      attn_out[((size_t)b * SS + srow) * DM + d] = acc[cb][v];
    }
  }
}

// =====================================================================
// Host-side orchestration.
// Workspace: q,k,v bf16 head-split (8 MB each) + attn_out fp32 (16 MB) = 40 MB.
// d_out: [output (B,S,D) fp32][attn_weights (B,H,S,S) fp32].
// =====================================================================
extern "C" void kernel_launch(void* const* d_in, const int* in_sizes, int n_in,
                              void* d_out, int out_size, void* d_ws, size_t ws_size,
                              hipStream_t stream) {
  (void)in_sizes; (void)n_in; (void)out_size; (void)ws_size;

  const float* Q  = (const float*)d_in[0];
  const float* K  = (const float*)d_in[1];
  const float* V  = (const float*)d_in[2];
  const float* Wq = (const float*)d_in[3];
  const float* bq = (const float*)d_in[4];
  const float* Wk = (const float*)d_in[5];
  const float* bk = (const float*)d_in[6];
  const float* Wv = (const float*)d_in[7];
  const float* bv = (const float*)d_in[8];
  const float* Wo = (const float*)d_in[9];
  const float* bo = (const float*)d_in[10];

  const size_t qkv_elems = (size_t)BB * NH * SS * DK;   // 4,194,304
  __bf16* q_ws = (__bf16*)d_ws;
  __bf16* k_ws = q_ws + qkv_elems;
  __bf16* v_ws = k_ws + qkv_elems;
  float*  ao_ws = (float*)(v_ws + qkv_elems);           // (B,S,D) fp32

  float* out    = (float*)d_out;                        // (B,S,D)
  float* attn_w = out + (size_t)NROW * DM;              // (B,H,S,S)

  const dim3 pgrid(DM / 64, NROW / 64);                 // (16, 64)
  proj_kernel<true ><<<pgrid, 128, 0, stream>>>(Q, Wq, bq, q_ws, nullptr);
  proj_kernel<true ><<<pgrid, 128, 0, stream>>>(K, Wk, bk, k_ws, nullptr);
  proj_kernel<true ><<<pgrid, 128, 0, stream>>>(V, Wv, bv, v_ws, nullptr);

  attn_scores_kernel<<<(BB * NH * (SS / 16)) / 4, 128, 0, stream>>>(q_ws, k_ws, attn_w);
  attn_pv_kernel   <<< BB * NH * (SS / 64),       128, 0, stream>>>(attn_w, v_ws, ao_ws);

  proj_kernel<false><<<pgrid, 128, 0, stream>>>(ao_ws, Wo, bo, nullptr, out);
}
